// Transformer2_33036888440860
// MI455X (gfx1250) — matrix operand order repaired and checked by reading, not compiled, not run on hardware
//
#include <hip/hip_runtime.h>
#include <hip/hip_bf16.h>

typedef __bf16 bf16_t;
typedef __attribute__((ext_vector_type(16))) __bf16 v16bf;
typedef __attribute__((ext_vector_type(8)))  __bf16 v8bf;
typedef __attribute__((ext_vector_type(8)))  float  v8f;
typedef __attribute__((ext_vector_type(4)))  unsigned int v4u;
typedef __attribute__((ext_vector_type(8)))  int v8i;
typedef __attribute__((ext_vector_type(4)))  int v4i;

#define WMMA_BF16(a, b, c) \
  __builtin_amdgcn_wmma_f32_16x16x32_bf16(false, (a), false, (b), (short)0, (c), false, false)

#define CAT16(lo, hi) \
  __builtin_shufflevector((lo), (hi), 0, 1, 2, 3, 4, 5, 6, 7, 8, 9, 10, 11, 12, 13, 14, 15)

__device__ __forceinline__ bf16_t f2bf(float f) {
  unsigned u = __float_as_uint(f);
  u = (u + 0x7FFFu + ((u >> 16) & 1u)) >> 16;  // round-to-nearest-even
  unsigned short h = (unsigned short)u;
  bf16_t r;
  __builtin_memcpy(&r, &h, sizeof(h));
  return r;
}

// ---------------------------------------------------------------------------
// q_in = queries + pe_q*32 ; k_in = q_in + pe_k*32 ; store f32 + bf16 copies
// ---------------------------------------------------------------------------
__global__ void prep_qk(const float* __restrict__ q, const float* __restrict__ peq,
                        const float* __restrict__ pek, float* __restrict__ qinf,
                        bf16_t* __restrict__ qinb, bf16_t* __restrict__ kinb, int total) {
  int i = blockIdx.x * blockDim.x + threadIdx.x;
  if (i >= total) return;
  int p = i & (1024 * 1024 - 1);  // index within [T,D]
  float qi = q[i] + peq[p] * 32.0f;
  float ki = qi + pek[p] * 32.0f;
  qinf[i] = qi;
  qinb[i] = f2bf(qi);
  kinb[i] = f2bf(ki);
}

// src f32 [K][N] -> dst bf16 [N][K]  (so WMMA B-frag loads are contiguous)
__global__ void transpose_bf(const float* __restrict__ src, bf16_t* __restrict__ dst,
                             int K, int N) {
  long i = (long)blockIdx.x * blockDim.x + threadIdx.x;
  long total = (long)K * N;
  if (i >= total) return;
  int n = (int)(i / K);
  int k = (int)(i - (long)n * K);
  dst[i] = f2bf(src[(long)k * N + n]);
}

// ---------------------------------------------------------------------------
// Generic bf16 WMMA GEMM:  C[M,N] = A[M,K] @ B[K,N],  Bt is [N][K] bf16.
// Block: 256 thr = 8 waves, block tile 256 rows x 64 cols; wave tile 32x64.
// B panel [64 cols x 64 k] staged in LDS by the Tensor Data Mover (wave 0
// issues TENSOR_LOAD_TO_LDS, drains TENSORcnt, barrier releases the readers).
// EPI 0: store bf16.  1: relu->bf16.  2: +Res(f32)->f32.  3: scatter Vt bf16.
// ---------------------------------------------------------------------------
template <int EPI>
__global__ __launch_bounds__(256) void gemm_wmma(
    const bf16_t* __restrict__ A, const bf16_t* __restrict__ Bt,
    float* __restrict__ Cf, bf16_t* __restrict__ Cb, const float* __restrict__ Res,
    int M, int N, int K) {
  constexpr int KB = 64;  // k-panel depth
  __shared__ __align__(16) bf16_t Bs[64 * KB];  // 8 KB: [col][k]

  const int lane = threadIdx.x & 31;
  const int wave = threadIdx.x >> 5;
  const int m16  = lane & 15;
  const int kh   = lane >> 4;
  const int row0 = blockIdx.y * 256 + wave * 32;
  const int col0 = blockIdx.x * 64;

  v8f acc[2][4];
#pragma unroll
  for (int rs = 0; rs < 2; ++rs)
#pragma unroll
    for (int nt = 0; nt < 4; ++nt)
#pragma unroll
      for (int r = 0; r < 8; ++r) acc[rs][nt][r] = 0.0f;

  const bf16_t* arow0 = A + (long)(row0 + m16) * K;
  const bf16_t* arow1 = A + (long)(row0 + 16 + m16) * K;
  const unsigned ldsb = (unsigned)(size_t)(&Bs[0]);  // low 32 bits = LDS offset

  for (int kb0 = 0; kb0 < K; kb0 += KB) {
    __syncthreads();  // previous panel fully consumed
    if (wave == 0) {
      // ---- Tensor DMA descriptor (ISA §8): 2-D tile of Bt[N][K] ----
      unsigned long long ga = (unsigned long long)(size_t)(Bt + (long)col0 * K + kb0);
      v4u g0;
      g0[0] = 1u;                                   // count=1, user mode
      g0[1] = ldsb;                                 // lds_addr
      g0[2] = (unsigned)ga;                         // global_addr[31:0]
      g0[3] = (unsigned)((ga >> 32) & 0x01FFFFFFu) | 0x80000000u;  // addr[56:32] | type=2
      v8i g1;
      g1[0] = (int)(1u << 16);                      // data_size=1 (2 bytes), no flags
      g1[1] = (int)(((unsigned)K & 0xFFFFu) << 16); // tensor_dim0[15:0]
      g1[2] = (int)((((unsigned)K >> 16) & 0xFFFFu) | (((unsigned)N & 0xFFFFu) << 16));
      g1[3] = (int)((((unsigned)N >> 16) & 0xFFFFu) | ((unsigned)KB << 16));  // tile_dim0
      g1[4] = 64;                                   // tile_dim1 = 64 cols (tile_dim2 = 0)
      g1[5] = K;                                    // tensor_dim0_stride[31:0]
      g1[6] = 0;
      g1[7] = 0;
      v4i gz4;
      gz4[0] = 0; gz4[1] = 0; gz4[2] = 0; gz4[3] = 0;  // groups 2/3 unused (2-D tile)
      v8i gz8;
#pragma unroll
      for (int z = 0; z < 8; ++z) gz8[z] = 0;
      __builtin_amdgcn_tensor_load_to_lds(g0, g1, gz4, gz4, gz8, 0);
      __builtin_amdgcn_s_wait_tensorcnt((short)0);
    }
    __syncthreads();  // panel visible to all waves

#pragma unroll
    for (int kk = 0; kk < KB; kk += 32) {
      const int kb = kb0 + kk;
      if (kb + 128 < K) {
        __builtin_prefetch(arow0 + kb + 128, 0, 1);
        __builtin_prefetch(arow1 + kb + 128, 0, 1);
      }
      // A-frags 16x32 (contiguous k order: lane half kh holds k = kh*16 .. +15)
      v16bf af0 = *(const v16bf*)(arow0 + kb + kh * 16);
      v16bf af1 = *(const v16bf*)(arow1 + kb + kh * 16);
#pragma unroll
      for (int nt = 0; nt < 4; ++nt) {
        // B-frag 32x16: lane n holds Bs[n][kk + kh*16 .. +15] (contiguous 32B)
        v16bf bfv = *(const v16bf*)(&Bs[(nt * 16 + m16) * KB + kk + kh * 16]);
        acc[0][nt] = WMMA_BF16(af0, bfv, acc[0][nt]);
        acc[1][nt] = WMMA_BF16(af1, bfv, acc[1][nt]);
      }
    }
  }

#pragma unroll
  for (int rs = 0; rs < 2; ++rs) {
#pragma unroll
    for (int nt = 0; nt < 4; ++nt) {
#pragma unroll
      for (int r = 0; r < 8; ++r) {
        int rowg = row0 + rs * 16 + r + 8 * kh;  // C layout: VGPR r -> M=r (+8 hi lanes)
        int colg = col0 + nt * 16 + m16;
        float v = acc[rs][nt][r];
        if (EPI == 0) {
          Cb[(long)rowg * N + colg] = f2bf(v);
        } else if (EPI == 1) {
          v = v > 0.0f ? v : 0.0f;
          Cb[(long)rowg * N + colg] = f2bf(v);
        } else if (EPI == 2) {
          long idx = (long)rowg * N + colg;
          Cf[idx] = v + Res[idx];
        } else {  // V transposed per head: Vt[(b*16+h)*64+dh][t]
          int b = rowg >> 10, t = rowg & 1023;
          int hh = colg >> 6, dh = colg & 63;
          Cb[(((long)(b * 16 + hh) * 64 + dh) << 10) + t] = f2bf(v);
        }
      }
    }
  }
}

// ---------------------------------------------------------------------------
// Flash attention: 1 wave per 16-query tile per (b,h).  T=1024, DH=64.
// scores = Q Kt / 8, key-mask + diag-blind -> NEG, online softmax,
// P@V via WMMA (P staged through LDS bf16), qmask + 1/l + residual epilogue.
// ---------------------------------------------------------------------------
__global__ __launch_bounds__(32) void attn_kernel(
    const bf16_t* __restrict__ Qb, const bf16_t* __restrict__ Kb,
    const bf16_t* __restrict__ Vt, const float* __restrict__ qinf,
    const int* __restrict__ qlens, const int* __restrict__ klens,
    float* __restrict__ res1f, bf16_t* __restrict__ res1b) {
  const int lane = threadIdx.x & 31;
  const int m16 = lane & 15, kh = lane >> 4;
  const int qt = blockIdx.x;           // 0..63 query tile
  const int b = blockIdx.y >> 4, h = blockIdx.y & 15;
  const int klen = klens[b], qlen = qlens[b];
  const float NEGV = -4294967295.0f;   // -2^32+1

  __shared__ __align__(16) bf16_t Pl[16][32];

  // Q A-frags for K-dims dh[0..31] and dh[32..63]
  const bf16_t* qrow = Qb + ((long)(b * 1024 + qt * 16 + m16) * 1024 + h * 64);
  v16bf aq0 = *(const v16bf*)(qrow + kh * 16);
  v16bf aq1 = *(const v16bf*)(qrow + 32 + kh * 16);

  float mrun[8], lrun[8];
  v8f acc[4];
#pragma unroll
  for (int r = 0; r < 8; ++r) { mrun[r] = -3.0e38f; lrun[r] = 0.0f; }
#pragma unroll
  for (int nt = 0; nt < 4; ++nt)
#pragma unroll
    for (int r = 0; r < 8; ++r) acc[nt][r] = 0.0f;

  const long kbase = (long)b * 1024 * 1024 + h * 64;
  const long vbase = (long)(b * 16 + h) * 64 * 1024;

  for (int kt = 0; kt < 32; ++kt) {
    const int j0 = kt * 32;
    // ---- scores: two 16x16 tiles over 32 keys (Bt of Q@K^T == K rows) ----
    v8f s0, s1;
#pragma unroll
    for (int r = 0; r < 8; ++r) { s0[r] = 0.0f; s1[r] = 0.0f; }
    {
      const bf16_t* kr0 = Kb + kbase + (long)(j0 + m16) * 1024;
      const bf16_t* kr1 = Kb + kbase + (long)(j0 + 16 + m16) * 1024;
      v16bf kb0a = *(const v16bf*)(kr0 + kh * 16);
      v16bf kb0b = *(const v16bf*)(kr0 + 32 + kh * 16);
      v16bf kb1a = *(const v16bf*)(kr1 + kh * 16);
      v16bf kb1b = *(const v16bf*)(kr1 + 32 + kh * 16);
      s0 = WMMA_BF16(aq0, kb0a, s0);
      s0 = WMMA_BF16(aq1, kb0b, s0);
      s1 = WMMA_BF16(aq0, kb1a, s1);
      s1 = WMMA_BF16(aq1, kb1b, s1);
    }
    const int c0 = j0 + m16, c1 = j0 + 16 + m16;
    float t0[8], t1[8];
#pragma unroll
    for (int r = 0; r < 8; ++r) {
      int qg = qt * 16 + r + 8 * kh;
      float v0 = s0[r] * 0.125f;                 // 1/sqrt(64)
      float v1 = s1[r] * 0.125f;
      if (c0 >= klen || c0 == qg) v0 = NEGV;     // key mask + diagonal blinding
      if (c1 >= klen || c1 == qg) v1 = NEGV;
      t0[r] = v0; t1[r] = v1;
    }
    // ---- online softmax (row = 16 lanes of one half; xor<=8 stays inside) ----
#pragma unroll
    for (int r = 0; r < 8; ++r) {
      float mx = fmaxf(t0[r], t1[r]);
#pragma unroll
      for (int m = 1; m <= 8; m <<= 1) mx = fmaxf(mx, __shfl_xor(mx, m, 32));
      float mnew = fmaxf(mrun[r], mx);
      float so = __expf(mrun[r] - mnew);
      float p0 = __expf(t0[r] - mnew);
      float p1 = __expf(t1[r] - mnew);
      float rs = p0 + p1;
#pragma unroll
      for (int m = 1; m <= 8; m <<= 1) rs += __shfl_xor(rs, m, 32);
      lrun[r] = lrun[r] * so + rs;
      mrun[r] = mnew;
#pragma unroll
      for (int nt = 0; nt < 4; ++nt) acc[nt][r] *= so;
      Pl[r + 8 * kh][m16]      = f2bf(p0);       // C-layout -> LDS
      Pl[r + 8 * kh][16 + m16] = f2bf(p1);
    }
    __syncthreads();
    // ---- P (16x32) A-frag from LDS, then acc += P @ V ----
    const bf16_t* prow = &Pl[m16][0];
    v16bf pa = *(const v16bf*)(prow + kh * 16);
#pragma unroll
    for (int nt = 0; nt < 4; ++nt) {
      const bf16_t* vrow = Vt + vbase + (long)(nt * 16 + m16) * 1024 + j0 + kh * 16;
      v16bf vb = *(const v16bf*)(vrow);
      acc[nt] = WMMA_BF16(pa, vb, acc[nt]);
    }
    __syncthreads();
  }

  // ---- epilogue: qmask, normalize, residual (+bf16 copy for FFN) ----
#pragma unroll
  for (int r = 0; r < 8; ++r) {
    int qg = qt * 16 + r + 8 * kh;
    float f = (qg < qlen) ? (1.0f / lrun[r]) : 0.0f;  // lrun >= 1 always
    long rowbase = ((long)b * 1024 + qg) * 1024 + h * 64;
#pragma unroll
    for (int nt = 0; nt < 4; ++nt) {
      long idx = rowbase + nt * 16 + m16;
      float v = acc[nt][r] * f + qinf[idx];
      res1f[idx] = v;
      res1b[idx] = f2bf(v);
    }
  }
}

// out[b*1024+d] = mean over T of res2
__global__ void mean_kernel(const float* __restrict__ res2, float* __restrict__ out) {
  int i = blockIdx.x * blockDim.x + threadIdx.x;  // 0..4095
  int b = i >> 10, d = i & 1023;
  const float* p = res2 + (long)b * 1024 * 1024 + d;
  float s = 0.0f;
  for (int t = 0; t < 1024; ++t) s += p[(long)t * 1024];
  out[i] = s * (1.0f / 1024.0f);
}

// ---------------------------------------------------------------------------
extern "C" void kernel_launch(void* const* d_in, const int* in_sizes, int n_in,
                              void* d_out, int out_size, void* d_ws, size_t ws_size,
                              hipStream_t stream) {
  const float* queries = (const float*)d_in[0];
  /* d_in[1] = keys: unused (reference source bug preserved) */
  const int* qlens = (const int*)d_in[2];
  const int* klens = (const int*)d_in[3];
  const float* peq = (const float*)d_in[4];
  const float* pek = (const float*)d_in[5];
  const float* Wq  = (const float*)d_in[6];
  const float* Wk  = (const float*)d_in[7];
  const float* Wv  = (const float*)d_in[8];
  const float* fw1 = (const float*)d_in[9];
  const float* fw2 = (const float*)d_in[10];
  float* out = (float*)d_out;

  const long BTD = 4L * 1024 * 1024;        // 4,194,304
  char* ws = (char*)d_ws;
  size_t off = 0;
  auto alloc = [&](size_t bytes) -> void* {
    void* p = ws + off;
    off += (bytes + 255) & ~(size_t)255;
    return p;
  };
  float*  qinf  = (float*)alloc(BTD * 4);
  bf16_t* qinb  = (bf16_t*)alloc(BTD * 2);
  bf16_t* kinb  = (bf16_t*)alloc(BTD * 2);
  bf16_t* wqt   = (bf16_t*)alloc(1024L * 1024 * 2);
  bf16_t* wkt   = (bf16_t*)alloc(1024L * 1024 * 2);
  bf16_t* wvt   = (bf16_t*)alloc(1024L * 1024 * 2);
  bf16_t* fw1t  = (bf16_t*)alloc(4096L * 1024 * 2);
  bf16_t* fw2t  = (bf16_t*)alloc(4096L * 1024 * 2);
  bf16_t* Qb    = (bf16_t*)alloc(BTD * 2);
  bf16_t* Kbm   = (bf16_t*)alloc(BTD * 2);
  bf16_t* Vtr   = (bf16_t*)alloc(BTD * 2);
  float*  res1f = (float*)alloc(BTD * 4);
  bf16_t* res1b = (bf16_t*)alloc(BTD * 2);
  bf16_t* hid   = (bf16_t*)alloc(4096L * 4096 * 2);
  float*  res2  = (float*)alloc(BTD * 4);

  // 1) PE + bf16 conversion
  prep_qk<<<(int)(BTD / 256), 256, 0, stream>>>(queries, peq, pek, qinf, qinb, kinb, (int)BTD);
  // 2) weight transposes: [K,N] f32 -> [N,K] bf16
  transpose_bf<<<4096, 256, 0, stream>>>(Wq, wqt, 1024, 1024);
  transpose_bf<<<4096, 256, 0, stream>>>(Wk, wkt, 1024, 1024);
  transpose_bf<<<4096, 256, 0, stream>>>(Wv, wvt, 1024, 1024);
  transpose_bf<<<16384, 256, 0, stream>>>(fw1, fw1t, 1024, 4096);  // -> [4096][1024]
  transpose_bf<<<16384, 256, 0, stream>>>(fw2, fw2t, 4096, 1024);  // -> [1024][4096]
  // 3) projections (block tile 256x64)
  gemm_wmma<0><<<dim3(16, 16), 256, 0, stream>>>(qinb, wqt, (float*)nullptr, Qb,
                                                 (const float*)nullptr, 4096, 1024, 1024);
  gemm_wmma<0><<<dim3(16, 16), 256, 0, stream>>>(kinb, wkt, (float*)nullptr, Kbm,
                                                 (const float*)nullptr, 4096, 1024, 1024);
  gemm_wmma<3><<<dim3(16, 16), 256, 0, stream>>>(Kbm, wvt, (float*)nullptr, Vtr,
                                                 (const float*)nullptr, 4096, 1024, 1024);
  // 4) attention + residual
  attn_kernel<<<dim3(64, 64), 32, 0, stream>>>(Qb, Kbm, Vtr, qinf, qlens, klens, res1f, res1b);
  // 5) FFN
  gemm_wmma<1><<<dim3(64, 16), 256, 0, stream>>>(res1b, fw1t, (float*)nullptr, hid,
                                                 (const float*)nullptr, 4096, 4096, 1024);
  gemm_wmma<2><<<dim3(16, 16), 256, 0, stream>>>(hid, fw2t, res2, (bf16_t*)nullptr,
                                                 res1f, 4096, 1024, 4096);
  // 6) mean over T
  mean_kernel<<<16, 256, 0, stream>>>(res2, out);
}
